// EdgePredictor_66400194396560
// MI455X (gfx1250) — compile-verified, matrix operand order checked by reading, not run on hardware
//
#include <hip/hip_runtime.h>
#include <hip/hip_bf16.h>
#include <math.h>

typedef __attribute__((ext_vector_type(2)))  float    v2f;
typedef __attribute__((ext_vector_type(8)))  float    v8f;
typedef __attribute__((ext_vector_type(16))) _Float16 v16h;

#define NEGV (-1.0e30f)
#define KNN 15
#define H 64

// ---------------------------------------------------------------------------
// utility: fill
// ---------------------------------------------------------------------------
__global__ void fill_kernel(float* __restrict__ p, float v, long n) {
  long i = (long)blockIdx.x * blockDim.x + threadIdx.x;
  if (i < n) p[i] = v;
}

// split f32 -> f16 hi + f16 lo (error compensation):  v ~= hi + lo, err ~2^-22
__global__ void split_hi_lo_kernel(const float* __restrict__ src,
                                   _Float16* __restrict__ hi,
                                   _Float16* __restrict__ lo, long n) {
  long i = (long)blockIdx.x * blockDim.x + threadIdx.x;
  if (i < n) {
    float v = src[i];
    _Float16 h = (_Float16)v;
    hi[i] = h;
    lo[i] = (_Float16)(v - (float)h);
  }
}

__global__ void cvt_f16_kernel(const float* __restrict__ src, _Float16* __restrict__ dst, long n) {
  long i = (long)blockIdx.x * blockDim.x + threadIdx.x;
  if (i < n) dst[i] = (_Float16)src[i];
}

// ---------------------------------------------------------------------------
// float atomic max via monotone integer mapping (init must be negative)
// ---------------------------------------------------------------------------
__device__ __forceinline__ void atomicMaxFloat(float* addr, float val) {
  if (val >= 0.0f) atomicMax((int*)addr, __float_as_int(val));
  else             atomicMin((unsigned int*)addr, __float_as_uint(val));
}

// ---------------------------------------------------------------------------
// 1) kNN: one 256-thread block per node; distances in LDS; 15 argmin passes
// ---------------------------------------------------------------------------
__global__ void __launch_bounds__(256) knn_kernel(const float* __restrict__ x,
                                                  int n, int* __restrict__ knn_col) {
  __shared__ float dist[4096];
  __shared__ float rv[256];
  __shared__ int   ri[256];
  int i = blockIdx.x;
  int tid = threadIdx.x;
  float xi0 = x[i * 3 + 0], xi1 = x[i * 3 + 1], xi2 = x[i * 3 + 2];
  float x2i = xi0 * xi0 + xi1 * xi1 + xi2 * xi2;
  for (int j = tid; j < n; j += 256) {
    float a = x[j * 3 + 0], b = x[j * 3 + 1], c = x[j * 3 + 2];
    float d = x2i + (a * a + b * b + c * c) - 2.0f * (xi0 * a + xi1 * b + xi2 * c);
    if (j == i) d += 1.0e30f;              // mask self-loop (matches eye*1e30)
    dist[j] = d;
  }
  __syncthreads();
  for (int r = 0; r < KNN; ++r) {
    float bv = 3.0e38f; int bi = 0x7fffffff;
    for (int j = tid; j < n; j += 256) {
      float v = dist[j];
      if (v < bv || (v == bv && j < bi)) { bv = v; bi = j; }
    }
    rv[tid] = bv; ri[tid] = bi;
    __syncthreads();
    for (int s = 128; s > 0; s >>= 1) {
      if (tid < s) {
        float ov = rv[tid + s]; int oi = ri[tid + s];
        if (ov < rv[tid] || (ov == rv[tid] && oi < ri[tid])) { rv[tid] = ov; ri[tid] = oi; }
      }
      __syncthreads();
    }
    if (tid == 0) {
      knn_col[i * KNN + r] = ri[0];
      dist[ri[0]] = 3.4e38f;               // exclude from later rounds
    }
    __syncthreads();
  }
}

// ---------------------------------------------------------------------------
// 2) edge theta + scatter-max:  theta = (x[row]-x[col]) @ W_theta^T + b_theta
// ---------------------------------------------------------------------------
__global__ void edge_theta_kernel(const float* __restrict__ x,
                                  const int* __restrict__ ei, int E,
                                  const int* __restrict__ knn_col, int Etot,
                                  const float* __restrict__ Wt,
                                  const float* __restrict__ bt,
                                  float* __restrict__ agg) {
  int e = blockIdx.x * blockDim.x + threadIdx.x;
  if (e >= Etot) return;
  int r, c;
  if (e < E) { r = ei[e]; c = ei[E + e]; }
  else       { int q = e - E; r = q / KNN; c = knn_col[q]; }
  float dx = x[r * 3 + 0] - x[c * 3 + 0];
  float dy = x[r * 3 + 1] - x[c * 3 + 1];
  float dz = x[r * 3 + 2] - x[c * 3 + 2];
  float* aggr = agg + (size_t)r * H;
  #pragma unroll 4
  for (int h = 0; h < H; ++h) {
    float th = bt[h] + dx * Wt[h * 3 + 0] + dy * Wt[h * 3 + 1] + dz * Wt[h * 3 + 2];
    atomicMaxFloat(&aggr[h], th);
  }
}

__global__ void fix_agg_kernel(float* __restrict__ agg, long n) {
  long i = (long)blockIdx.x * blockDim.x + threadIdx.x;
  if (i < n) { float v = agg[i]; agg[i] = (v <= NEGV * 0.5f) ? 0.0f : v; }
}

// ---------------------------------------------------------------------------
// 3) small exact-f32 WMMA GEMM: out[M,64] = A[M,64] @ Wt[64,64]^T (+bias)
// ---------------------------------------------------------------------------
__global__ void __launch_bounds__(32) small_gemm_wmma(const float* __restrict__ A,
                                                      const float* __restrict__ Wt,
                                                      const float* __restrict__ bias,
                                                      float* __restrict__ out) {
  const int K = H, Nn = H;
  int m0 = blockIdx.x * 16;
  int lane = threadIdx.x;
  int lh = lane >> 4;            // 0|1 : K-half (A: K=k,k+1 vs k+2,k+3)
  int ll = lane & 15;
  v8f c[4];
  #pragma unroll
  for (int t = 0; t < 4; ++t) c[t] = (v8f){0,0,0,0,0,0,0,0};

  for (int k = 0; k < K; k += 4) {
    int kk = k + 2 * lh;
    v2f a;
    a.x = A[(size_t)(m0 + ll) * K + kk];
    a.y = A[(size_t)(m0 + ll) * K + kk + 1];
    #pragma unroll
    for (int t = 0; t < 4; ++t) {
      int n = t * 16 + ll;
      v2f b;
      b.x = Wt[(size_t)n * K + kk];       // B[k][n] = Wt[n][k]
      b.y = Wt[(size_t)n * K + kk + 1];
      c[t] = __builtin_amdgcn_wmma_f32_16x16x4_f32(false, a, false, b,
                                                   (short)0, c[t], false, false);
    }
  }
  #pragma unroll
  for (int t = 0; t < 4; ++t) {
    float bv = bias ? bias[t * 16 + ll] : 0.0f;
    #pragma unroll
    for (int r = 0; r < 8; ++r) {
      int row = m0 + r + 8 * lh;          // C/D: VGPR r -> M=r (lo) / r+8 (hi)
      out[(size_t)row * Nn + t * 16 + ll] = c[t][r] + bv;
    }
  }
}

// ---------------------------------------------------------------------------
// 4) attention on original edges
// ---------------------------------------------------------------------------
__global__ void att_dot_kernel(const float* __restrict__ q, const float* __restrict__ kf,
                               const int* __restrict__ ei, int E, float* __restrict__ att) {
  int e = blockIdx.x * blockDim.x + threadIdx.x;
  if (e >= E) return;
  int r = ei[e], c = ei[E + e];
  const float4* qa = (const float4*)(q + (size_t)r * H);
  const float4* ka = (const float4*)(kf + (size_t)c * H);
  float s = 0.0f;
  #pragma unroll
  for (int i = 0; i < H / 4; ++i) {
    float4 a = qa[i], b = ka[i];
    s += a.x * b.x + a.y * b.y + a.z * b.z + a.w * b.w;
  }
  att[e] = s;
}

__global__ void att_max_kernel(const float* __restrict__ att, const int* __restrict__ ei,
                               int E, float* __restrict__ m) {
  int e = blockIdx.x * blockDim.x + threadIdx.x;
  if (e >= E) return;
  atomicMaxFloat(&m[ei[e]], att[e]);
}

__global__ void att_expsum_kernel(const float* __restrict__ att, const int* __restrict__ ei,
                                  int E, const float* __restrict__ m,
                                  float* __restrict__ attex, float* __restrict__ s) {
  int e = blockIdx.x * blockDim.x + threadIdx.x;
  if (e >= E) return;
  int r = ei[e];
  float ev = expf(att[e] - m[r]);
  attex[e] = ev;
  atomicAdd(&s[r], ev);
}

// scatter softmax values into dense S; scatter ones into TRANSPOSED adjacency At
__global__ void scatter_SA_kernel(const float* __restrict__ attex, const float* __restrict__ s,
                                  const int* __restrict__ ei, int E,
                                  float* __restrict__ S, float* __restrict__ At, int N) {
  int e = blockIdx.x * blockDim.x + threadIdx.x;
  if (e >= E) return;
  int r = ei[e], c = ei[E + e];
  atomicAdd(&S[(size_t)r * N + c], attex[e] / s[r]);
  atomicAdd(&At[(size_t)c * N + r], 1.0f);         // At[n][k] = A[k][n]
}

// ---------------------------------------------------------------------------
// 5) big split-precision f16 WMMA GEMM (v_wmma_f32_16x16x32_f16)
//    C[M,N](f32) = Ah@B (+ Al@B) (+ Ah@Bl)   with B given transposed row-major:
//      Bt[n][k] = B[k][n]
//    mode 1: C = (Ah+Al) @ B          (Bl unused)      -> T = S @ A
//    mode 2: C = Ah@Bh + Al@Bh + Ah@Bl  (drop lo*lo)   -> out = T @ S^T
//    128x128 block tile, 8 waves (2x4), wave = 64x32 = 4x2 WMMA tiles, K-slab 32
// ---------------------------------------------------------------------------
#define BT 128
#define KS 32
#define STR 40   // LDS row stride in halfs (32 + 8 pad -> conflict-free b128 reads)

union HFrag { v16h v; uint4 q[2]; };

__global__ void __launch_bounds__(256) big_gemm_f16(const _Float16* __restrict__ Ah,
                                                    const _Float16* __restrict__ Al,
                                                    const _Float16* __restrict__ Bth,
                                                    const _Float16* __restrict__ Btl,
                                                    float* __restrict__ C, int N, int mode) {
  __shared__ _Float16 AsH[BT * STR];
  __shared__ _Float16 AsL[BT * STR];
  __shared__ _Float16 BsH[BT * STR];
  __shared__ _Float16 BsL[BT * STR];
  const int tm0 = blockIdx.y * BT, tn0 = blockIdx.x * BT;
  const int t = threadIdx.x;
  const int wave = t >> 5, lane = t & 31;
  const int wm = wave >> 2;                         // 0..1 : 64-row slice
  const int wn = wave & 3;                          // 0..3 : 32-col slice
  const int lh = lane >> 4, ll = lane & 15;

  v8f acc[4][2];
  #pragma unroll
  for (int mt = 0; mt < 4; ++mt)
    #pragma unroll
    for (int nt = 0; nt < 2; ++nt) acc[mt][nt] = (v8f){0,0,0,0,0,0,0,0};

  // stage a 128x32 f16 tile (rows row0.., cols k0..k0+31) into padded LDS
  auto stage = [&](const _Float16* __restrict__ src, int row0, int k0,
                   _Float16* __restrict__ dst, bool pf) {
    #pragma unroll
    for (int i = 0; i < 2; ++i) {
      int idx = t + i * 256;                        // 0..511 b128 chunks
      int rr = idx >> 2, c4 = idx & 3;              // 4 x 16B per 32-half row
      const _Float16* gp = src + (size_t)(row0 + rr) * N + k0;
      uint4 v = *(const uint4*)(gp + c4 * 8);
      *(uint4*)&dst[rr * STR + c4 * 8] = v;
      if (pf) __builtin_prefetch((const void*)(gp + KS), 0, 1);  // next K-slab -> global_prefetch_b8
    }
  };

  for (int k0 = 0; k0 < N; k0 += KS) {
    bool pf = (k0 + KS) < N;
    __syncthreads();
    stage(Ah,  tm0, k0, AsH, pf);
    stage(Al,  tm0, k0, AsL, pf);
    stage(Bth, tn0, k0, BsH, pf);
    if (mode == 2) stage(Btl, tn0, k0, BsL, pf);
    __syncthreads();

    // f16 fragments, ISA 16-bit A-layout: lane half lh -> K = 8*lh+0..7, 16+8*lh+0..7
    HFrag ah[4], al[4], bh[2], bl[2];
    #pragma unroll
    for (int mt = 0; mt < 4; ++mt) {
      int base = (wm * 64 + mt * 16 + ll) * STR + 8 * lh;
      ah[mt].q[0] = *(const uint4*)&AsH[base];
      ah[mt].q[1] = *(const uint4*)&AsH[base + 16];
      al[mt].q[0] = *(const uint4*)&AsL[base];
      al[mt].q[1] = *(const uint4*)&AsL[base + 16];
    }
    #pragma unroll
    for (int nt = 0; nt < 2; ++nt) {
      int base = (wn * 32 + nt * 16 + ll) * STR + 8 * lh;
      bh[nt].q[0] = *(const uint4*)&BsH[base];
      bh[nt].q[1] = *(const uint4*)&BsH[base + 16];
      if (mode == 2) {
        bl[nt].q[0] = *(const uint4*)&BsL[base];
        bl[nt].q[1] = *(const uint4*)&BsL[base + 16];
      }
    }
    #pragma unroll
    for (int mt = 0; mt < 4; ++mt)
      #pragma unroll
      for (int nt = 0; nt < 2; ++nt) {
        acc[mt][nt] = __builtin_amdgcn_wmma_f32_16x16x32_f16(false, ah[mt].v, false, bh[nt].v,
                                                             (short)0, acc[mt][nt], false, false);
        acc[mt][nt] = __builtin_amdgcn_wmma_f32_16x16x32_f16(false, al[mt].v, false, bh[nt].v,
                                                             (short)0, acc[mt][nt], false, false);
        if (mode == 2)
          acc[mt][nt] = __builtin_amdgcn_wmma_f32_16x16x32_f16(false, ah[mt].v, false, bl[nt].v,
                                                               (short)0, acc[mt][nt], false, false);
      }
  }

  #pragma unroll
  for (int mt = 0; mt < 4; ++mt)
    #pragma unroll
    for (int nt = 0; nt < 2; ++nt) {
      int col = tn0 + wn * 32 + nt * 16 + ll;
      #pragma unroll
      for (int r = 0; r < 8; ++r) {
        int row = tm0 + wm * 64 + mt * 16 + r + 8 * lh;
        C[(size_t)row * N + col] = acc[mt][nt][r];
      }
    }
}

// ---------------------------------------------------------------------------
// host-side orchestration
// ---------------------------------------------------------------------------
extern "C" void kernel_launch(void* const* d_in, const int* in_sizes, int n_in,
                              void* d_out, int out_size, void* d_ws, size_t ws_size,
                              hipStream_t stream) {
  const float* x      = (const float*)d_in[0];
  const int*   ei     = (const int*)d_in[1];        // [2,E] row-major flat
  const float* Wtheta = (const float*)d_in[2];
  const float* btheta = (const float*)d_in[3];
  const float* Wphi   = (const float*)d_in[4];
  const float* bphi   = (const float*)d_in[5];
  const float* Wq     = (const float*)d_in[6];
  const float* Wk     = (const float*)d_in[7];
  float* out = (float*)d_out;

  const int N = in_sizes[0] / 3;
  const int E = in_sizes[1] / 2;
  const int Etot = E + N * KNN;

  // workspace layout (256B aligned slabs)
  char* ws = (char*)d_ws;
  size_t off = 0;
  auto take = [&](size_t bytes) { size_t o = off; off += (bytes + 255) & ~(size_t)255; return o; };
  float*    F1    = (float*)   (ws + take((size_t)N * N * 4));  // S (f32), then T (f32)
  _Float16* Shi   = (_Float16*)(ws + take((size_t)N * N * 2));
  _Float16* Slo   = (_Float16*)(ws + take((size_t)N * N * 2));
  _Float16* Gx    = (_Float16*)(ws + take((size_t)N * N * 2));  // At (f16), then T_hi
  _Float16* Gy    = (_Float16*)(ws + take((size_t)N * N * 2));  // T_lo
  float*    agg   = (float*)   (ws + take((size_t)N * H * 4));
  float*    feats = (float*)   (ws + take((size_t)N * H * 4));
  float*    q     = (float*)   (ws + take((size_t)N * H * 4));
  float*    kf    = (float*)   (ws + take((size_t)N * H * 4));
  float*    att   = (float*)   (ws + take((size_t)E * 4));
  float*    attex = (float*)   (ws + take((size_t)E * 4));
  float*    mrow  = (float*)   (ws + take((size_t)N * 4));
  float*    srow  = (float*)   (ws + take((size_t)N * 4));
  int*      knn_c = (int*)     (ws + take((size_t)N * KNN * 4));

  auto cdiv = [](long a, long b) { return (int)((a + b - 1) / b); };

  // init
  fill_kernel<<<cdiv((long)N * H, 256), 256, 0, stream>>>(agg, NEGV, (long)N * H);
  fill_kernel<<<cdiv((long)N, 256), 256, 0, stream>>>(mrow, NEGV, (long)N);
  fill_kernel<<<cdiv((long)N, 256), 256, 0, stream>>>(srow, 0.0f, (long)N);
  fill_kernel<<<cdiv((long)N * N, 256), 256, 0, stream>>>(F1, 0.0f, (long)N * N);   // dense S
  fill_kernel<<<cdiv((long)N * N, 256), 256, 0, stream>>>(out, 0.0f, (long)N * N);  // dense A^T built here

  // kNN edges
  knn_kernel<<<N, 256, 0, stream>>>(x, N, knn_c);

  // devconv
  edge_theta_kernel<<<cdiv(Etot, 256), 256, 0, stream>>>(x, ei, E, knn_c, Etot, Wtheta, btheta, agg);
  fix_agg_kernel<<<cdiv((long)N * H, 256), 256, 0, stream>>>(agg, (long)N * H);
  small_gemm_wmma<<<N / 16, 32, 0, stream>>>(agg, Wphi, bphi, feats);

  // attention
  small_gemm_wmma<<<N / 16, 32, 0, stream>>>(feats, Wq, nullptr, q);
  small_gemm_wmma<<<N / 16, 32, 0, stream>>>(feats, Wk, nullptr, kf);
  att_dot_kernel<<<cdiv(E, 256), 256, 0, stream>>>(q, kf, ei, E, att);
  att_max_kernel<<<cdiv(E, 256), 256, 0, stream>>>(att, ei, E, mrow);
  att_expsum_kernel<<<cdiv(E, 256), 256, 0, stream>>>(att, ei, E, mrow, attex, srow);
  scatter_SA_kernel<<<cdiv(E, 256), 256, 0, stream>>>(attex, srow, ei, E, F1, out, N);

  // precision-split operands for f16 WMMA
  split_hi_lo_kernel<<<cdiv((long)N * N, 256), 256, 0, stream>>>(F1, Shi, Slo, (long)N * N);
  cvt_f16_kernel<<<cdiv((long)N * N, 256), 256, 0, stream>>>(out, Gx, (long)N * N);  // A^T exact in f16

  // A_s = (S @ A) @ S^T  via split-f16 WMMA GEMMs
  dim3 g(N / BT, N / BT);
  big_gemm_f16<<<g, 256, 0, stream>>>(Shi, Slo, Gx, nullptr, F1, N, 1);   // T = S @ A   (B^T = At)
  split_hi_lo_kernel<<<cdiv((long)N * N, 256), 256, 0, stream>>>(F1, Gx, Gy, (long)N * N);
  big_gemm_f16<<<g, 256, 0, stream>>>(Gx, Gy, Shi, Slo, out, N, 2);       // out = T @ S^T (B^T = S)
}
